// KroneckerMixer_82102594830394
// MI455X (gfx1250) — matrix operand order repaired and checked by reading, not compiled, not run on hardware
//
#include <hip/hip_runtime.h>

#define INV_TEMP 5.0f
#define SINK_ITERS 20
#define APITCH 40   // f16 pitch for 128x32 LDS tiles (80B rows, 16B-aligned chunks)
#define WPITCH 72   // f16 pitch for 64x64 LDS tiles (144B rows, 16B-aligned chunks)

typedef __attribute__((ext_vector_type(16))) _Float16 v16h;
typedef __attribute__((ext_vector_type(8)))  _Float16 v8h;
typedef __attribute__((ext_vector_type(8)))  float    v8f;
typedef __attribute__((ext_vector_type(4)))  unsigned int u32x4;

__device__ __forceinline__ v16h make_v16h(v8h lo, v8h hi) {
    v16h r;
#pragma unroll
    for (int i = 0; i < 8; ++i) { r[i] = lo[i]; r[i + 8] = hi[i]; }
    return r;
}

// ---------------- Sinkhorn on A (1024x1024, f32, L2-resident) ----------------

__global__ void k_init_logA(const float* __restrict__ Alog, float* __restrict__ logA) {
    int base = blockIdx.x * 1024 + threadIdx.x;
#pragma unroll
    for (int e = 0; e < 4; ++e) {
        int j = base + e * 256;
        logA[j] = Alog[j] * INV_TEMP;
    }
}

// one block per row: subtract row logsumexp (row cached in LDS)
__global__ void __launch_bounds__(256) k_rowpass(float* __restrict__ logA) {
    __shared__ float row[1024];
    __shared__ float red[256];
    const int t = threadIdx.x;
    float* gp = logA + (size_t)blockIdx.x * 1024;
    for (int j = t; j < 1024; j += 256) row[j] = gp[j];
    __syncthreads();
    float m = -3.4e38f;
    for (int j = t; j < 1024; j += 256) m = fmaxf(m, row[j]);
    red[t] = m; __syncthreads();
    for (int s = 128; s > 0; s >>= 1) { if (t < s) red[t] = fmaxf(red[t], red[t + s]); __syncthreads(); }
    m = red[0]; __syncthreads();
    float sa = 0.f;
    for (int j = t; j < 1024; j += 256) sa += __expf(row[j] - m);
    red[t] = sa; __syncthreads();
    for (int s = 128; s > 0; s >>= 1) { if (t < s) red[t] += red[t + s]; __syncthreads(); }
    float lse = m + __logf(red[0]);
    for (int j = t; j < 1024; j += 256) gp[j] = row[j] - lse;
}

// per-column partial (max,sumexp) over 128-row chunks; grid (4, 8)
__global__ void k_colreduce(const float* __restrict__ logA, float* __restrict__ part) {
    int col = blockIdx.x * 256 + threadIdx.x;
    const float* p = logA + (size_t)blockIdx.y * 128 * 1024 + col;
    float m = -3.4e38f;
    for (int i = 0; i < 128; ++i) m = fmaxf(m, p[i * 1024]);
    float s = 0.f;
    for (int i = 0; i < 128; ++i) s += __expf(p[i * 1024] - m);
    float* o = part + ((size_t)blockIdx.y * 1024 + col) * 2;
    o[0] = m; o[1] = s;
}

__global__ void k_colcombine(const float* __restrict__ part, float* __restrict__ lse) {
    int col = blockIdx.x * 256 + threadIdx.x;
    float M = -3.4e38f;
#pragma unroll
    for (int c = 0; c < 8; ++c) M = fmaxf(M, part[(c * 1024 + col) * 2]);
    float S = 0.f;
#pragma unroll
    for (int c = 0; c < 8; ++c) S += part[(c * 1024 + col) * 2 + 1] * __expf(part[(c * 1024 + col) * 2] - M);
    lse[col] = M + __logf(S);
}

// subtract col-LSE; on final iteration emit exp() as f16 A matrix
__global__ void k_colsub(float* __restrict__ logA, const float* __restrict__ lse,
                         _Float16* __restrict__ A16, int finalize) {
    int base = blockIdx.x * 1024 + threadIdx.x;
#pragma unroll
    for (int e = 0; e < 4; ++e) {
        int j = base + e * 256;
        float v = logA[j] - lse[j & 1023];
        if (finalize) A16[j] = (_Float16)__expf(v);
        else          logA[j] = v;
    }
}

// ------- per-n: W_logits -> sinkhorn in LDS -> exp -> X_n @ W_n via WMMA -------

__global__ void __launch_bounds__(256) k_local(const float* __restrict__ x,
                                               const float* __restrict__ W1,
                                               const float* __restrict__ WV,
                                               _Float16* __restrict__ L) {
    __shared__ __attribute__((aligned(16))) float    wl[4096];          // 64x64 logits
    __shared__ __attribute__((aligned(16))) _Float16 Wt[64 * WPITCH];   // W^T (col-major W)
    __shared__ __attribute__((aligned(16))) _Float16 Xs[64 * WPITCH];   // X_n row-major f16
    __shared__ float w1s[8];
    const int t = threadIdx.x, n = blockIdx.x;

    if (t < 8) w1s[t] = W1[n * 8 + t];
    __syncthreads();

    // W_logits = sum_k W1[n,k] * WV[k,:,:]  (scaled by 1/temp)
#pragma unroll
    for (int e = 0; e < 16; ++e) {
        int idx = t + e * 256;
        float acc = 0.f;
#pragma unroll
        for (int k = 0; k < 8; ++k) acc += w1s[k] * WV[k * 4096 + idx];
        wl[idx] = acc * INV_TEMP;
    }
    __syncthreads();

    // 20 Sinkhorn iterations fully in LDS
    for (int it = 0; it < SINK_ITERS; ++it) {
        if (t < 64) {                                   // row pass (reduce over e)
            float* rp = wl + t * 64;
            float m = -3.4e38f;
            for (int c = 0; c < 64; ++c) m = fmaxf(m, rp[c]);
            float s = 0.f;
            for (int c = 0; c < 64; ++c) s += __expf(rp[c] - m);
            float l = m + __logf(s);
            for (int c = 0; c < 64; ++c) rp[c] -= l;
        }
        __syncthreads();
        if (t < 64) {                                   // col pass (reduce over d)
            float m = -3.4e38f;
            for (int d = 0; d < 64; ++d) m = fmaxf(m, wl[d * 64 + t]);
            float s = 0.f;
            for (int d = 0; d < 64; ++d) s += __expf(wl[d * 64 + t] - m);
            float l = m + __logf(s);
            for (int d = 0; d < 64; ++d) wl[d * 64 + t] -= l;
        }
        __syncthreads();
    }

    // W^T into LDS as f16 (B-operand wants contiguous-in-K at fixed column)
#pragma unroll
    for (int e = 0; e < 16; ++e) {
        int idx = t + e * 256;
        int d = idx >> 6, c = idx & 63;
        Wt[c * WPITCH + d] = (_Float16)__expf(wl[idx]);
    }
    // X_n tile: X[b][k] = x[b*65536 + n*64 + k], f16 row-major
    {
        int b = t >> 2, c0 = (t & 3) * 16;
        const float* xp = x + (size_t)b * 65536 + n * 64 + c0;
#pragma unroll
        for (int j = 0; j < 16; ++j) Xs[b * WPITCH + c0 + j] = (_Float16)xp[j];
    }
    __syncthreads();

    // Y_n(64x64) = X_n @ W : 16 tiles of 16x16, 8 waves x 2 tiles, K = 2x32
    const int wave = t >> 5, lane = t & 31, hlf = lane >> 4, lrow = lane & 15;
    const int tm = wave & 3, tnb = (wave >> 2) * 2;
    const int arow = tm * 16 + lrow;
    v16h a0 = make_v16h(*(const v8h*)(Xs + arow * WPITCH + hlf * 8),
                        *(const v8h*)(Xs + arow * WPITCH + 16 + hlf * 8));
    v16h a1 = make_v16h(*(const v8h*)(Xs + arow * WPITCH + 32 + hlf * 8),
                        *(const v8h*)(Xs + arow * WPITCH + 48 + hlf * 8));
#pragma unroll
    for (int ti = 0; ti < 2; ++ti) {
        int col = (tnb + ti) * 16 + lrow;
        v16h b0 = make_v16h(*(const v8h*)(Wt + col * WPITCH + hlf * 8),
                            *(const v8h*)(Wt + col * WPITCH + 16 + hlf * 8));
        v16h b1 = make_v16h(*(const v8h*)(Wt + col * WPITCH + 32 + hlf * 8),
                            *(const v8h*)(Wt + col * WPITCH + 48 + hlf * 8));
        v8f c = {};
        c = __builtin_amdgcn_wmma_f32_16x16x32_f16(false, a0, false, b0, (short)0, c, false, false);
        c = __builtin_amdgcn_wmma_f32_16x16x32_f16(false, a1, false, b1, (short)0, c, false, false);
        _Float16* op = L + (size_t)n * 4096 + col;      // L[n][b*64+o]
#pragma unroll
        for (int v = 0; v < 8; ++v) {
            int row = tm * 16 + v + hlf * 8;            // b
            op[row * 64] = (_Float16)c[v];
        }
    }
}

// ---- x_global = A(1024x1024) @ L(1024x4096): double-buffered WMMA GEMM ----
// A tile staged with GLOBAL_LOAD_ASYNC_TO_LDS_B128 (ASYNCcnt), B tile staged
// via VGPR prefetch + in-LDS transpose; one barrier per k-step.

__global__ void __launch_bounds__(256) k_global(const _Float16* __restrict__ A16,
                                                const _Float16* __restrict__ L,
                                                float* __restrict__ out) {
    __shared__ __attribute__((aligned(16))) _Float16 As[2][128 * APITCH];
    __shared__ __attribute__((aligned(16))) _Float16 Bt[2][128 * APITCH];
    const int t = threadIdx.x;
    const int mbase = blockIdx.x * 128, nbase = blockIdx.y * 128;
    const int wave = t >> 5, lane = t & 31, hlf = lane >> 4, lrow = lane & 15;
    const int wm = wave & 3, wn = wave >> 2;            // wave tile: 32 rows x 64 cols

    v8f acc[2][4];
#pragma unroll
    for (int i = 0; i < 2; ++i)
#pragma unroll
        for (int j = 0; j < 4; ++j) acc[i][j] = (v8f){};

    const int ar = t >> 1, ac0 = (t & 1) * 16;          // A loader: 32B per thread
    const int bk = (t >> 4) * 2, bc0 = (t & 15) * 8;    // B loader: 2 rows x 8 cols

    // async DMA of this thread's 32B slice of the A tile (global -> LDS)
    auto loadA = [&](int kt, int buf) {
        unsigned lds0 = (unsigned)(size_t)(&As[buf][ar * APITCH + ac0]);
        unsigned long long g0 =
            (unsigned long long)(size_t)(A16 + (size_t)(mbase + ar) * 1024 + kt * 32 + ac0);
        asm volatile("global_load_async_to_lds_b128 %0, %1, off"
                     :: "v"(lds0), "v"(g0) : "memory");
        asm volatile("global_load_async_to_lds_b128 %0, %1, off offset:16"
                     :: "v"(lds0), "v"(g0) : "memory");
    };

    v8h bv0, bv1;                                       // B(kt) staged in VGPRs
    loadA(0, 0);
    {
        const _Float16* r0 = L + (size_t)bk * 4096 + nbase + bc0;
        bv0 = *(const v8h*)r0;
        bv1 = *(const v8h*)(r0 + 4096);
    }

    for (int kt = 0; kt < 32; ++kt) {
        const int buf = kt & 1;
        // transpose-store B(kt) VGPRs -> Bt[buf]  (Bt[col][k])
#pragma unroll
        for (int j = 0; j < 8; ++j) {
            union { _Float16 h[2]; unsigned int u; } p;
            p.h[0] = bv0[j]; p.h[1] = bv1[j];
            *(unsigned int*)(&Bt[buf][(bc0 + j) * APITCH + bk]) = p.u;
        }
        // A(kt) async portion of this wave must be resident before the barrier
        asm volatile("s_wait_asynccnt 0x0" ::: "memory");
        __syncthreads();

        // issue next tile's loads so they overlap this tile's WMMA block
        if (kt < 31) {
            loadA(kt + 1, buf ^ 1);
            const _Float16* r0 = L + (size_t)((kt + 1) * 32 + bk) * 4096 + nbase + bc0;
            bv0 = *(const v8h*)r0;
            bv1 = *(const v8h*)(r0 + 4096);
        }

        v16h af[2], bf[4];
#pragma unroll
        for (int ti = 0; ti < 2; ++ti) {
            int row = wm * 32 + ti * 16 + lrow;
            af[ti] = make_v16h(*(const v8h*)(&As[buf][row * APITCH + hlf * 8]),
                               *(const v8h*)(&As[buf][row * APITCH + 16 + hlf * 8]));
        }
#pragma unroll
        for (int tj = 0; tj < 4; ++tj) {
            int col = wn * 64 + tj * 16 + lrow;
            bf[tj] = make_v16h(*(const v8h*)(&Bt[buf][col * APITCH + hlf * 8]),
                               *(const v8h*)(&Bt[buf][col * APITCH + 16 + hlf * 8]));
        }
#pragma unroll
        for (int ti = 0; ti < 2; ++ti)
#pragma unroll
            for (int tj = 0; tj < 4; ++tj)
                acc[ti][tj] = __builtin_amdgcn_wmma_f32_16x16x32_f16(
                    false, af[ti], false, bf[tj], (short)0, acc[ti][tj], false, false);
        // no second barrier: double buffering + next iteration's barrier
        // separates any reuse of this buffer
    }

    // C[m, nc] -> out[b, m*64 + o], nc = b*64+o (each tile stays within one b)
#pragma unroll
    for (int ti = 0; ti < 2; ++ti)
#pragma unroll
        for (int tj = 0; tj < 4; ++tj) {
            int nc = nbase + wn * 64 + tj * 16 + lrow;
            int b = nc >> 6, o = nc & 63;
            float* op = out + (size_t)b * 65536 + o;
#pragma unroll
            for (int v = 0; v < 8; ++v) {
                int m = mbase + wm * 32 + ti * 16 + v + hlf * 8;
                op[m * 64] = acc[ti][tj][v];
            }
        }
}

// ------------------------------- launcher ------------------------------------

extern "C" void kernel_launch(void* const* d_in, const int* in_sizes, int n_in,
                              void* d_out, int out_size, void* d_ws, size_t ws_size,
                              hipStream_t stream) {
    (void)in_sizes; (void)n_in; (void)out_size; (void)ws_size;
    const float* x     = (const float*)d_in[0];  // [64, 65536]
    const float* Alog  = (const float*)d_in[1];  // [1024, 1024]
    const float* W1    = (const float*)d_in[2];  // [1024, 8]
    const float* WV    = (const float*)d_in[3];  // [8, 64, 64]
    float* out = (float*)d_out;                  // [64, 65536]

    char* ws = (char*)d_ws;
    float*    logA    = (float*)(ws);                                    // 4 MB
    float*    colPart = (float*)(ws + (4u << 20));                       // 64 KB
    float*    colLse  = (float*)(ws + (4u << 20) + (64u << 10));         // 4 KB
    _Float16* A16     = (_Float16*)(ws + (4u << 20) + (128u << 10));     // 2 MB
    _Float16* Lws     = (_Float16*)(ws + (6u << 20) + (128u << 10));     // 8 MB

    k_init_logA<<<1024, 256, 0, stream>>>(Alog, logA);
    for (int it = 0; it < SINK_ITERS; ++it) {
        k_rowpass<<<1024, 256, 0, stream>>>(logA);
        k_colreduce<<<dim3(4, 8), 256, 0, stream>>>(logA, colPart);
        k_colcombine<<<4, 256, 0, stream>>>(colPart, colLse);
        k_colsub<<<1024, 256, 0, stream>>>(logA, colLse, A16, it == SINK_ITERS - 1);
    }
    k_local<<<1024, 256, 0, stream>>>(x, W1, WV, Lws);
    k_global<<<dim3(8, 32), 256, 0, stream>>>(A16, Lws, out);
}
